// HierarchicalGPT_66279935311942
// MI455X (gfx1250) — matrix operand order repaired and checked by reading, not compiled, not run on hardware
//
#include <hip/hip_runtime.h>
#include <hip/hip_bf16.h>

typedef __attribute__((ext_vector_type(16))) _Float16 v16h;
typedef __attribute__((ext_vector_type(8)))  float    v8f;
typedef _Float16 f16;

static constexpr int Bb = 2, T = 1024, Cc = 384, Hh = 4, Dh = 96;
static constexpr int NTOK = Bb * T;            // 2048
static constexpr int C3 = 3 * Cc;              // 1152
static constexpr int Ee = 8, FF = 1536;
static constexpr float EPS_LN = 1e-5f;
static constexpr float ATT_SCALE = 0.1020620726159658f; // 1/sqrt(96)

// ---------------------------------------------------------------------------
// gfx1250 async global->LDS copy (16 bytes per lane), tracked by ASYNCcnt.
// LDS raw address = low 32 bits of the flat shared-memory address (ISA 10.2).
// ---------------------------------------------------------------------------
__device__ __forceinline__ void async_g2l_b128(const f16* gsrc, f16* ldst) {
  unsigned lds = (unsigned)(unsigned long long)ldst;
  asm volatile("global_load_async_to_lds_b128 %0, %1, off"
               :: "v"(lds), "v"(gsrc) : "memory");
}
__device__ __forceinline__ void wait_async0() {
  asm volatile("s_wait_asynccnt 0" ::: "memory");
}

// ---------------------------------------------------------------------------
// f32 -> f16 conversion
// ---------------------------------------------------------------------------
__global__ void k_cvt_f16(const float* __restrict__ a, f16* __restrict__ o, int n) {
  int i = blockIdx.x * 256 + threadIdx.x;
  if (i < n) o[i] = (f16)a[i];
}

// ---------------------------------------------------------------------------
// LayerNorm over last dim C (block = 128 threads, one row per block).
// ---------------------------------------------------------------------------
__global__ void k_layernorm(const float* __restrict__ x, const float* __restrict__ g,
                            const float* __restrict__ b, float* __restrict__ y,
                            f16* __restrict__ y16, int C) {
  const int row = blockIdx.x;
  const float* xr = x + (long long)row * C;
  __shared__ float red[128];
  float s = 0.f;
  for (int c = threadIdx.x; c < C; c += 128) s += xr[c];
  red[threadIdx.x] = s; __syncthreads();
  for (int o = 64; o > 0; o >>= 1) { if (threadIdx.x < o) red[threadIdx.x] += red[threadIdx.x + o]; __syncthreads(); }
  const float mean = red[0] / (float)C; __syncthreads();
  s = 0.f;
  for (int c = threadIdx.x; c < C; c += 128) { float d = xr[c] - mean; s += d * d; }
  red[threadIdx.x] = s; __syncthreads();
  for (int o = 64; o > 0; o >>= 1) { if (threadIdx.x < o) red[threadIdx.x] += red[threadIdx.x + o]; __syncthreads(); }
  const float rstd = rsqrtf(red[0] / (float)C + EPS_LN);
  for (int c = threadIdx.x; c < C; c += 128) {
    float v = (xr[c] - mean) * rstd * g[c] + b[c];
    y[(long long)row * C + c] = v;
    if (y16) y16[(long long)row * C + c] = (f16)v;
  }
}

// ---------------------------------------------------------------------------
// WMMA GEMM v3: block = 128 threads = 4 waves, 64x32 macro-tile per block.
// Wave w computes rows [mbase + w*16, +16) x 32 columns (2 accumulators,
// A-operand reused across the WMMA pair).
// A tile (64x32) staged via async global->LDS b128 copies.
// B tile staged TRANSPOSED in LDS as Bt[n][k] so each lane's WMMA B operand
// is two contiguous ds_load_b128 reads.
//   btrans=1: B source is N x K row-major -> contiguous async copy.
//   btrans=0: B source is K x N row-major -> coalesced b128 load + b16 scatter.
// Epilogue modes:
//   0 PLAIN : C = acc (+bias[n]);  optional f16 mirror
//   1 CAUSAL: C = (n<=m)? acc*scale : -3e38
//   2 GELU  : Ch = (f16) gelu(acc + bias[n])      (exact erf gelu)
//   3 WACC  : Cf += rowW[m*rwStride] * (acc + bias[n])
// ---------------------------------------------------------------------------
__global__ void k_gemm_wmma(const f16* __restrict__ A, const f16* __restrict__ Bm,
                            float* __restrict__ Cf, f16* __restrict__ Ch,
                            int K, int lda, int ldb, int ldc,
                            long long sA, long long sB, long long sC,
                            int btrans, int mode, float scale,
                            const float* __restrict__ bias,
                            const float* __restrict__ rowW, int rwStride) {
  __shared__ f16 As[64][40];   // [m][k]: 64 rows x 32 halves (+8 pad, 80B rows)
  __shared__ f16 Bt[32][40];   // [n][k]: 32 cols x 32 halves (+8 pad)
  const int tid  = threadIdx.x;
  const int lane = tid & 31;
  const int w    = tid >> 5;          // wave id 0..3
  const int lhi  = (lane >> 4) & 1;   // lane half
  const int lm   = lane & 15;
  const int nbase = blockIdx.x * 32;
  const int mbase = blockIdx.y * 64;
  const int bz    = blockIdx.z;
  A  += (long long)bz * sA;
  Bm += (long long)bz * sB;

  v8f acc0, acc1;
#pragma unroll
  for (int r = 0; r < 8; ++r) { acc0[r] = 0.f; acc1[r] = 0.f; }

  const int arow  = tid >> 1;          // 0..63
  const int acol  = (tid & 1) * 16;    // 0 or 16 (16 halves = 32B per thread)
  const int brow  = tid >> 2;          // 0..31 (k for non-trans, n for trans)
  const int bcol8 = (tid & 3) * 8;     // 0,8,16,24 (8 halves = 16B per thread)

  for (int k0 = 0; k0 < K; k0 += 32) {
    // ---- stage A tile: two async b128 per thread ----
    {
      const f16* src = A + (long long)(mbase + arow) * lda + k0 + acol;
      async_g2l_b128(src,     &As[arow][acol]);
      async_g2l_b128(src + 8, &As[arow][acol + 8]);
      if (k0 + 32 < K) __builtin_prefetch(src + 32, 0, 1);
    }
    // ---- stage B tile into Bt[n][k] ----
    if (btrans) {
      // source row n holds contiguous k: straight async copy
      const f16* src = Bm + (long long)(nbase + brow) * ldb + k0 + bcol8;
      async_g2l_b128(src, &Bt[brow][bcol8]);
      if (k0 + 32 < K) __builtin_prefetch(src + 32, 0, 1);
    } else {
      // source row k holds contiguous n: coalesced load, scatter to Bt
      const f16* src = Bm + (long long)(k0 + brow) * ldb + nbase + bcol8;
      f16 tmp[8];
#pragma unroll
      for (int j = 0; j < 8; ++j) tmp[j] = src[j];
#pragma unroll
      for (int j = 0; j < 8; ++j) Bt[bcol8 + j][brow] = tmp[j];
      if (k0 + 32 < K) __builtin_prefetch(src + (long long)32 * ldb, 0, 1);
    }
    wait_async0();
    __syncthreads();

    // ---- build WMMA operands from LDS (contiguous b128 reads) ----
    v16h a, b0, b1;
    {
      const f16* ap = &As[w * 16 + lm][lhi * 8];
#pragma unroll
      for (int e = 0; e < 8; ++e) { a[e] = ap[e]; a[e + 8] = ap[16 + e]; }
      const f16* bp0 = &Bt[lm][8 * lhi];         // K = 8*lhi + e, then 16 + 8*lhi + e
      const f16* bp1 = &Bt[16 + lm][8 * lhi];
#pragma unroll
      for (int e = 0; e < 8; ++e) { b0[e] = bp0[e]; b0[e + 8] = bp0[16 + e]; b1[e] = bp1[e]; b1[e + 8] = bp1[16 + e]; }
    }
    acc0 = __builtin_amdgcn_wmma_f32_16x16x32_f16(false, a, false, b0, (short)0, acc0, false, false);
    acc1 = __builtin_amdgcn_wmma_f32_16x16x32_f16(false, a, false, b1, (short)0, acc1, false, false);
    __syncthreads();
  }

  // ---- epilogue ----
#pragma unroll
  for (int r = 0; r < 8; ++r) {
    const int m = mbase + w * 16 + r + 8 * lhi;
#pragma unroll
    for (int half = 0; half < 2; ++half) {
      const int n = nbase + half * 16 + lm;
      float v = half ? acc1[r] : acc0[r];
      const long long oi = (long long)bz * sC + (long long)m * ldc + n;
      if (mode == 0) {
        if (bias) v += bias[n];
        Cf[oi] = v;
        if (Ch) Ch[oi] = (f16)v;
      } else if (mode == 1) {
        Cf[oi] = (n <= m) ? v * scale : -3.0e38f;
      } else if (mode == 2) {
        v += bias[n];
        Ch[oi] = (f16)(0.5f * v * (1.0f + erff(v * 0.7071067811865475f)));
      } else {
        float wgt = rowW[(long long)m * rwStride];
        Cf[oi] += wgt * (v + bias[n]);
      }
    }
  }
}

// ---------------------------------------------------------------------------
// Split qkv, apply rotary to q,k; emit f16 q/k/v in (B,H,T,D) layout,
// f32 v, and Qp/Kp = elu(x*scale)+1 for the linear-attention path.
// ---------------------------------------------------------------------------
__global__ void k_rotary(const float* __restrict__ qkv, float* __restrict__ vf,
                         f16* __restrict__ q16, f16* __restrict__ k16, f16* __restrict__ v16p,
                         float* __restrict__ Qp, float* __restrict__ Kp) {
  int i = blockIdx.x * 256 + threadIdx.x;
  if (i >= Bb * Hh * T * Dh) return;
  int d = i % Dh; int t = (i / Dh) % T; int h = (i / (Dh * T)) % Hh; int b = i / (Dh * T * Hh);
  const long long row = (long long)(b * T + t) * C3;
  const int col = h * Dh + d;
  float qv = qkv[row + col];
  float kv = qkv[row + Cc + col];
  float vv = qkv[row + 2 * Cc + col];
  int j = d % 48;
  float inv = powf(10000.f, -((float)(2 * j) / 96.f));
  float fr = (float)t * inv;
  float cs = cosf(fr), sn = sinf(fr);
  float qo = (d < 48) ? -qkv[row + h * Dh + d + 48]      : qkv[row + h * Dh + d - 48];
  float ko = (d < 48) ? -qkv[row + Cc + h * Dh + d + 48] : qkv[row + Cc + h * Dh + d - 48];
  float qr = qv * cs + qo * sn;
  float kr = kv * cs + ko * sn;
  const long long o = (long long)((b * Hh + h) * T + t) * Dh + d;
  q16[o] = (f16)qr; k16[o] = (f16)kr;
  vf[o] = vv; v16p[o] = (f16)vv;
  float zq = qr * ATT_SCALE, zk = kr * ATT_SCALE;
  Qp[o] = (zq > 0.f) ? zq + 1.f : expf(zq);
  Kp[o] = (zk > 0.f) ? zk + 1.f : expf(zk);
}

// ---------------------------------------------------------------------------
// Linear attention, chunked (16-token chunks, 64 chunks per (b,h)).
// ---------------------------------------------------------------------------
__global__ void k_chunk_sums(const float* __restrict__ Kp, const float* __restrict__ vf,
                             float* __restrict__ Skv, float* __restrict__ Sk) {
  const int cb = blockIdx.x;              // bh*64 + c
  const int bh = cb >> 6, c = cb & 63;
  __shared__ float kS[16 * 96], vS[16 * 96];
  const long long base = ((long long)bh * T + c * 16) * Dh;
  for (int i = threadIdx.x; i < 16 * 96; i += 256) { kS[i] = Kp[base + i]; vS[i] = vf[base + i]; }
  __syncthreads();
  float* so = Skv + (long long)cb * 96 * 96;
  for (int i = threadIdx.x; i < 96 * 96; i += 256) {
    int d = i / 96, e = i % 96;
    float s = 0.f;
#pragma unroll
    for (int t2 = 0; t2 < 16; ++t2) s += kS[t2 * 96 + d] * vS[t2 * 96 + e];
    so[i] = s;
  }
  if (threadIdx.x < 96) {
    float s = 0.f;
#pragma unroll
    for (int t2 = 0; t2 < 16; ++t2) s += kS[t2 * 96 + threadIdx.x];
    Sk[(long long)cb * 96 + threadIdx.x] = s;
  }
}

__global__ void k_chunk_prefix(float* __restrict__ Skv, float* __restrict__ Sk) {
  const int bh = blockIdx.x;
  for (int i = threadIdx.x; i < 96 * 96; i += 256) {
    float run = 0.f;
    for (int c = 0; c < 64; ++c) {
      long long o = (long long)(bh * 64 + c) * 9216 + i;
      float t2 = Skv[o]; Skv[o] = run; run += t2;
    }
  }
  if (threadIdx.x < 96) {
    float run = 0.f;
    for (int c = 0; c < 64; ++c) {
      long long o = (long long)(bh * 64 + c) * 96 + threadIdx.x;
      float t2 = Sk[o]; Sk[o] = run; run += t2;
    }
  }
}

__global__ void k_linattn(const float* __restrict__ Qp, const float* __restrict__ Kp,
                          const float* __restrict__ vf, const float* __restrict__ Skv,
                          const float* __restrict__ Sk, float* __restrict__ ylin) {
  const int cb = blockIdx.x; const int bh = cb >> 6, c = cb & 63;
  __shared__ float qS[1536], kS[1536], vS[1536], A[256], den[16];
  const long long base = ((long long)bh * T + c * 16) * Dh;
  for (int i = threadIdx.x; i < 1536; i += 256) { qS[i] = Qp[base + i]; kS[i] = Kp[base + i]; vS[i] = vf[base + i]; }
  __syncthreads();
  {
    int t2 = threadIdx.x >> 4, s2 = threadIdx.x & 15;
    float a = 0.f;
    for (int d = 0; d < 96; ++d) a += qS[t2 * 96 + d] * kS[s2 * 96 + d];
    A[threadIdx.x] = a;
  }
  __syncthreads();
  const float* Pk = Sk + (long long)cb * 96;
  if (threadIdx.x < 16) {
    int t2 = threadIdx.x;
    float s = 0.f;
    for (int d = 0; d < 96; ++d) s += qS[t2 * 96 + d] * Pk[d];
    for (int s2 = 0; s2 <= t2; ++s2) s += A[t2 * 16 + s2];
    den[t2] = s + 1e-4f;
  }
  __syncthreads();
  const float* Pkv = Skv + (long long)cb * 9216;
  for (int i = threadIdx.x; i < 1536; i += 256) {
    int t2 = i / 96, e = i % 96;
    float num = 0.f;
    for (int d = 0; d < 96; ++d) num += qS[t2 * 96 + d] * Pkv[d * 96 + e];
    for (int s2 = 0; s2 <= t2; ++s2) num += A[t2 * 16 + s2] * vS[s2 * 96 + e];
    ylin[base + i] = num / den[t2];
  }
}

// ---------------------------------------------------------------------------
// Row softmax (length T), masked entries are -3e38; emit f16 probabilities.
// ---------------------------------------------------------------------------
__global__ void k_softmax(const float* __restrict__ S, f16* __restrict__ P) {
  const int t = blockIdx.x;
  const float* r = S + (long long)t * T;
  __shared__ float red[256];
  float m = -3.4e38f;
  for (int s = threadIdx.x; s < T; s += 256) m = fmaxf(m, r[s]);
  red[threadIdx.x] = m; __syncthreads();
  for (int o = 128; o > 0; o >>= 1) { if (threadIdx.x < o) red[threadIdx.x] = fmaxf(red[threadIdx.x], red[threadIdx.x + o]); __syncthreads(); }
  m = red[0]; __syncthreads();
  float sum = 0.f;
  for (int s = threadIdx.x; s < T; s += 256) sum += expf(r[s] - m);
  red[threadIdx.x] = sum; __syncthreads();
  for (int o = 128; o > 0; o >>= 1) { if (threadIdx.x < o) red[threadIdx.x] += red[threadIdx.x + o]; __syncthreads(); }
  const float invs = 1.0f / red[0];
  f16* pr = P + (long long)t * T;
  for (int s = threadIdx.x; s < T; s += 256) pr[s] = (f16)(expf(r[s] - m) * invs);
}

// ---------------------------------------------------------------------------
// Gate blend + (B,H,T,D) -> (B,T,C) merge.
// ---------------------------------------------------------------------------
__global__ void k_combine(const float* __restrict__ ysharp, const float* __restrict__ ylinn,
                          const float* __restrict__ focus, float* __restrict__ att0) {
  int i = blockIdx.x * 256 + threadIdx.x;
  if (i >= NTOK * Cc) return;
  int cc = i % Cc; long long bt = i / Cc;
  int b = (int)(bt / T), t = (int)(bt % T);
  int h = cc / Dh, d = cc % Dh;
  float a = 1.f / (1.f + expf(-focus[0]));
  long long o = (long long)((b * Hh + h) * T + t) * Dh + d;
  att0[i] = a * ysharp[o] + (1.f - a) * ylinn[o];
}

// ---------------------------------------------------------------------------
// Hierarchical block shuffle: block mixing + flip + monarch + residual.
// ---------------------------------------------------------------------------
__global__ void k_hbs(const float* __restrict__ xin, const float* __restrict__ blocks,
                      const float* __restrict__ wrow, const float* __restrict__ wcol,
                      const float* __restrict__ alpha, const float* __restrict__ bias,
                      const float* __restrict__ resid, float* __restrict__ hout) {
  const int n = blockIdx.x;
  __shared__ float row[400], bout[384], tmp[400], wc[400], wr[400];
  const float* xr = xin + (long long)n * Cc;
  for (int i = threadIdx.x; i < 400; i += 128) {
    row[i] = (i < Cc) ? xr[i] : 0.f;
    wc[i] = wcol[i]; wr[i] = wrow[i];
  }
  __syncthreads();
  for (int o = threadIdx.x; o < Cc; o += 128) {          // block mixing
    int g = o >> 4, cidx = o & 15;
    const float* bw = blocks + g * 256;
    float s = 0.f;
#pragma unroll
    for (int b2 = 0; b2 < 16; ++b2) s += row[g * 16 + b2] * bw[b2 * 16 + cidx];
    bout[o] = s;
  }
  for (int o = threadIdx.x; o < 400; o += 128) {         // monarch stage 1
    int i2 = o / 20, k2 = o % 20;
    float s = 0.f;
#pragma unroll
    for (int j = 0; j < 20; ++j) s += row[i2 * 20 + j] * wc[j * 20 + k2];
    tmp[o] = s;
  }
  __syncthreads();
  const float al = alpha[0];
  const float* rr = resid + (long long)n * Cc;
  float* ho = hout + (long long)n * Cc;
  for (int o = threadIdx.x; o < Cc; o += 128) {          // monarch stage 2 + combine
    int i2 = o / 20, l2 = o % 20;
    float mon = 0.f;
#pragma unroll
    for (int k2 = 0; k2 < 20; ++k2) mon += tmp[i2 * 20 + k2] * wr[l2 * 20 + k2];
    ho[o] = rr[o] + bout[383 - o] + al * mon + bias[o];
  }
}

// ---------------------------------------------------------------------------
// MoE gate: logits -> clip -> top-2 -> softmax -> dense mask wfull (N x 8).
// ---------------------------------------------------------------------------
__global__ void k_gate(const float* __restrict__ xn2, const float* __restrict__ gw,
                       float* __restrict__ wfull) {
  const int n = blockIdx.x;
  __shared__ float part[128][8];
  const float* xr = xn2 + (long long)n * Cc;
  float acc[8] = {0.f, 0.f, 0.f, 0.f, 0.f, 0.f, 0.f, 0.f};
  for (int c = threadIdx.x; c < Cc; c += 128) {
    float xv = xr[c];
    const float* g = gw + c * 8;
#pragma unroll
    for (int e = 0; e < 8; ++e) acc[e] += xv * g[e];
  }
#pragma unroll
  for (int e = 0; e < 8; ++e) part[threadIdx.x][e] = acc[e];
  __syncthreads();
  if (threadIdx.x < 8) {
    int e = threadIdx.x;
    float s = 0.f;
    for (int i = 0; i < 128; ++i) s += part[i][e];
    part[0][e] = fminf(20.f, fmaxf(-20.f, s));
  }
  __syncthreads();
  if (threadIdx.x == 0) {
    float m1 = -1e30f, m2 = -1e30f; int i1 = 0, i2 = 0;
    for (int e = 0; e < 8; ++e) {
      float v = part[0][e];
      if (v > m1) { m2 = m1; i2 = i1; m1 = v; i1 = e; }
      else if (v > m2) { m2 = v; i2 = e; }
    }
    float p2 = 1.f / (1.f + expf(m1 - m2));
    float p1 = 1.f - p2;
    float* wrp = wfull + (long long)n * 8;
    for (int e = 0; e < 8; ++e) wrp[e] = 0.f;
    wrp[i1] = p1; wrp[i2] = p2;
  }
}

__global__ void k_add(const float* __restrict__ a, const float* __restrict__ b,
                      float* __restrict__ o, int n) {
  int i = blockIdx.x * 256 + threadIdx.x;
  if (i < n) o[i] = a[i] + b[i];
}

// ---------------------------------------------------------------------------
extern "C" void kernel_launch(void* const* d_in, const int* in_sizes, int n_in,
                              void* d_out, int out_size, void* d_ws, size_t ws_size,
                              hipStream_t stream) {
  (void)in_sizes; (void)n_in; (void)out_size; (void)ws_size;
  const float* x        = (const float*)d_in[0];
  const float* ln1_g    = (const float*)d_in[1];
  const float* ln1_b    = (const float*)d_in[2];
  const float* ln2_g    = (const float*)d_in[3];
  const float* ln2_b    = (const float*)d_in[4];
  const float* qkv_w    = (const float*)d_in[5];
  const float* qkv_b    = (const float*)d_in[6];
  const float* pre_g    = (const float*)d_in[7];
  const float* pre_b    = (const float*)d_in[8];
  const float* blocks   = (const float*)d_in[9];
  const float* wrow     = (const float*)d_in[10];
  const float* wcol     = (const float*)d_in[11];
  const float* alpha    = (const float*)d_in[12];
  const float* pbias    = (const float*)d_in[13];
  const float* an_g     = (const float*)d_in[14];
  const float* an_b     = (const float*)d_in[15];
  const float* on_g     = (const float*)d_in[16];
  const float* on_b     = (const float*)d_in[17];
  const float* focus    = (const float*)d_in[18];
  const float* gate_w   = (const float*)d_in[19];
  const float* moe_w1   = (const float*)d_in[20];
  const float* moe_b1   = (const float*)d_in[21];
  const float* moe_w2   = (const float*)d_in[22];
  const float* moe_b2   = (const float*)d_in[23];
  float* out = (float*)d_out;

  char* p = (char*)d_ws;
  auto alloc = [&](size_t bytes) -> char* {
    char* r = p; p += (bytes + 255) & ~(size_t)255; return r;
  };
  const long long BHTD = (long long)Bb * Hh * T * Dh;           // 786432
  f16*   qkvw16 = (f16*)alloc((size_t)Cc * C3 * 2);
  f16*   w1_16  = (f16*)alloc((size_t)Ee * Cc * FF * 2);
  f16*   w2_16  = (f16*)alloc((size_t)Ee * FF * Cc * 2);
  float* xn1    = (float*)alloc((size_t)NTOK * Cc * 4);
  f16*   xn1h   = (f16*)alloc((size_t)NTOK * Cc * 2);
  float* qkvb   = (float*)alloc((size_t)NTOK * C3 * 4);
  float* vf     = (float*)alloc((size_t)BHTD * 4);
  f16*   q16    = (f16*)alloc((size_t)BHTD * 2);
  f16*   k16    = (f16*)alloc((size_t)BHTD * 2);
  f16*   v16p   = (f16*)alloc((size_t)BHTD * 2);
  float* Qp     = (float*)alloc((size_t)BHTD * 4);
  float* Kp     = (float*)alloc((size_t)BHTD * 4);
  float* Skv    = (float*)alloc((size_t)Bb * Hh * 64 * 9216 * 4);
  float* Sk     = (float*)alloc((size_t)Bb * Hh * 64 * 96 * 4);
  float* ylin   = (float*)alloc((size_t)BHTD * 4);
  float* ylinn  = (float*)alloc((size_t)BHTD * 4);
  float* scores = (float*)alloc((size_t)T * T * 4);
  f16*   probs  = (f16*)alloc((size_t)T * T * 2);
  float* ysharp = (float*)alloc((size_t)BHTD * 4);
  float* att0   = (float*)alloc((size_t)NTOK * Cc * 4);
  float* att1   = (float*)alloc((size_t)NTOK * Cc * 4);
  float* hbs_in = (float*)alloc((size_t)NTOK * Cc * 4);
  float* hfull  = (float*)alloc((size_t)NTOK * Cc * 4);
  float* xn2    = (float*)alloc((size_t)NTOK * Cc * 4);
  f16*   xn2h   = (f16*)alloc((size_t)NTOK * Cc * 2);
  float* wfull  = (float*)alloc((size_t)NTOK * 8 * 4);
  f16*   hmoe16 = (f16*)alloc((size_t)NTOK * FF * 2);
  float* outmoe = (float*)alloc((size_t)NTOK * Cc * 4);

  // weight conversions (deterministic every call)
  k_cvt_f16<<<(Cc * C3 + 255) / 256, 256, 0, stream>>>(qkv_w, qkvw16, Cc * C3);
  k_cvt_f16<<<(Ee * Cc * FF + 255) / 256, 256, 0, stream>>>(moe_w1, w1_16, Ee * Cc * FF);
  k_cvt_f16<<<(Ee * FF * Cc + 255) / 256, 256, 0, stream>>>(moe_w2, w2_16, Ee * FF * Cc);

  // LN1 and QKV projection:  M=2048, N=1152, K=384
  k_layernorm<<<NTOK, 128, 0, stream>>>(x, ln1_g, ln1_b, xn1, xn1h, Cc);
  k_gemm_wmma<<<dim3(C3 / 32, NTOK / 64, 1), 128, 0, stream>>>(
      xn1h, qkvw16, qkvb, nullptr, Cc, Cc, C3, C3, 0, 0, 0, 0, 0, 1.f, qkv_b, nullptr, 0);

  // rotary split
  k_rotary<<<(int)((BHTD + 255) / 256), 256, 0, stream>>>(qkvb, vf, q16, k16, v16p, Qp, Kp);

  // linear attention (chunked)
  k_chunk_sums<<<Bb * Hh * 64, 256, 0, stream>>>(Kp, vf, Skv, Sk);
  k_chunk_prefix<<<Bb * Hh, 256, 0, stream>>>(Skv, Sk);
  k_linattn<<<Bb * Hh * 64, 256, 0, stream>>>(Qp, Kp, vf, Skv, Sk, ylin);
  k_layernorm<<<Bb * Hh * T, 128, 0, stream>>>(ylin, an_g, an_b, ylinn, nullptr, Dh);

  // softmax attention per (b,h): scores -> softmax -> P@V
  for (int bh = 0; bh < Bb * Hh; ++bh) {
    const long long off = (long long)bh * T * Dh;
    k_gemm_wmma<<<dim3(T / 32, T / 64, 1), 128, 0, stream>>>(
        q16 + off, k16 + off, scores, nullptr, Dh, Dh, Dh, T, 0, 0, 0,
        1 /*btrans*/, 1 /*causal*/, ATT_SCALE, nullptr, nullptr, 0);
    k_softmax<<<T, 256, 0, stream>>>(scores, probs);
    k_gemm_wmma<<<dim3(Dh / 32, T / 64, 1), 128, 0, stream>>>(
        probs, v16p + off, ysharp + off, nullptr, T, T, Dh, Dh, 0, 0, 0,
        0, 0, 1.f, nullptr, nullptr, 0);
  }

  // blend paths, merge heads, out-norm, HBS projection (+residual)
  k_combine<<<(NTOK * Cc + 255) / 256, 256, 0, stream>>>(ysharp, ylinn, focus, att0);
  k_layernorm<<<NTOK, 128, 0, stream>>>(att0, on_g, on_b, att1, nullptr, Cc);
  k_layernorm<<<NTOK, 128, 0, stream>>>(att1, pre_g, pre_b, hbs_in, nullptr, Cc);
  k_hbs<<<NTOK, 128, 0, stream>>>(hbs_in, blocks, wrow, wcol, alpha, pbias, x, hfull);

  // MoE: LN2 -> gate -> dense experts with top-2 mask weighting
  k_layernorm<<<NTOK, 128, 0, stream>>>(hfull, ln2_g, ln2_b, xn2, xn2h, Cc);
  k_gate<<<NTOK, 128, 0, stream>>>(xn2, gate_w, wfull);
  hipMemsetAsync(outmoe, 0, (size_t)NTOK * Cc * 4, stream);
  for (int e = 0; e < Ee; ++e) {
    k_gemm_wmma<<<dim3(FF / 32, NTOK / 64, 1), 128, 0, stream>>>(
        xn2h, w1_16 + (size_t)e * Cc * FF, nullptr, hmoe16, Cc, Cc, FF, FF, 0, 0, 0,
        0, 2 /*gelu*/, 1.f, moe_b1 + e * FF, nullptr, 0);
    k_gemm_wmma<<<dim3(Cc / 32, NTOK / 64, 1), 128, 0, stream>>>(
        hmoe16, w2_16 + (size_t)e * FF * Cc, outmoe, nullptr, FF, FF, Cc, Cc, 0, 0, 0,
        0, 3 /*wacc*/, 1.f, moe_b2 + e * Cc, wfull + e, 8);
  }

  // final residual
  k_add<<<(NTOK * Cc + 255) / 256, 256, 0, stream>>>(hfull, outmoe, out, NTOK * Cc);
}
